// TransformerEncoderLayer_2473901163040
// MI455X (gfx1250) — compile-verified, matrix operand order checked by reading, not run on hardware
//
#include <hip/hip_runtime.h>
#include <math.h>

// ---------------------------------------------------------------------------
// Types for CDNA5 WMMA / TDM
// ---------------------------------------------------------------------------
typedef __attribute__((ext_vector_type(16))) __bf16 v16bf;
typedef __attribute__((ext_vector_type(8)))  float  v8f;
typedef __attribute__((ext_vector_type(4)))  unsigned int v4u;
typedef __attribute__((ext_vector_type(8)))  int    v8i;
typedef __attribute__((ext_vector_type(4)))  int    v4i;

#define S_DIM 4096
#define B_DIM 8
#define E_DIM 1024
#define H_DIM 16
#define HD_DIM 64
#define FF_DIM 4096
#define MROWS (S_DIM * B_DIM)   // 32768

// ---------------------------------------------------------------------------
// Transpose f32 [K,N] -> bf16 [N,K]  (weights are transposed ONCE so GEMM
// B-tiles become K-contiguous; eliminates per-block LDS scatter)
// ---------------------------------------------------------------------------
__global__ __launch_bounds__(256) void transpose_bf16_kernel(
    const float* __restrict__ in, __bf16* __restrict__ out, int K, int N) {
  __shared__ float tile[32][33];
  const int n0 = blockIdx.x * 32;
  const int k0 = blockIdx.y * 32;
  const int tx = threadIdx.x & 31;
  const int ty = threadIdx.x >> 5;  // 0..7
#pragma unroll
  for (int i = 0; i < 32; i += 8)
    tile[ty + i][tx] = in[(size_t)(k0 + ty + i) * N + n0 + tx];
  __syncthreads();
#pragma unroll
  for (int i = 0; i < 32; i += 8)
    out[(size_t)(n0 + ty + i) * K + k0 + tx] = (__bf16)tile[tx][ty + i];
}

// ---------------------------------------------------------------------------
// LayerNorm over E=1024, one block per row, output bf16
// ---------------------------------------------------------------------------
__global__ __launch_bounds__(256) void ln_bf16_kernel(
    const float* __restrict__ x, const float* __restrict__ g,
    const float* __restrict__ b, __bf16* __restrict__ out) {
  __shared__ float red[256];
  const int tid = threadIdx.x;
  const size_t row = blockIdx.x;
  const float* xr = x + row * (size_t)E_DIM;

  float4 v = ((const float4*)xr)[tid];
  float s = v.x + v.y + v.z + v.w;
  red[tid] = s;
  __syncthreads();
  for (int o = 128; o > 0; o >>= 1) {
    if (tid < o) red[tid] += red[tid + o];
    __syncthreads();
  }
  const float mean = red[0] * (1.0f / E_DIM);
  __syncthreads();

  float dx0 = v.x - mean, dx1 = v.y - mean, dx2 = v.z - mean, dx3 = v.w - mean;
  red[tid] = dx0 * dx0 + dx1 * dx1 + dx2 * dx2 + dx3 * dx3;
  __syncthreads();
  for (int o = 128; o > 0; o >>= 1) {
    if (tid < o) red[tid] += red[tid + o];
    __syncthreads();
  }
  const float var = red[0] * (1.0f / E_DIM);
  const float rs = rsqrtf(var + 1e-5f);

  const int c = tid * 4;
  __bf16* orow = out + row * (size_t)E_DIM;
  orow[c + 0] = (__bf16)(dx0 * rs * g[c + 0] + b[c + 0]);
  orow[c + 1] = (__bf16)(dx1 * rs * g[c + 1] + b[c + 1]);
  orow[c + 2] = (__bf16)(dx2 * rs * g[c + 2] + b[c + 2]);
  orow[c + 3] = (__bf16)(dx3 * rs * g[c + 3] + b[c + 3]);
}

// ---------------------------------------------------------------------------
// TDM: async-load a 32(K) x 128(rows) bf16 tile of a [rows, K] row-major
// tensor into LDS with padded rows (64B data + 16B pad -> 80B stride).
// D# layout per CDNA5 ISA.
// ---------------------------------------------------------------------------
#define BM 128
#define BN 128
#define BK 32
#define LDT 40  // LDS row stride in bf16 elems (80B; 64B tile row + 16B pad)

__device__ __forceinline__ void tdm_load_tile(unsigned lds_off,
                                              const __bf16* gptr,
                                              int tensor_rows, int K) {
  const unsigned long long ga = (unsigned long long)(uintptr_t)gptr;
  v4u g0;
  g0[0] = 1u;                                   // count=1, user mode
  g0[1] = lds_off;                              // LDS byte address
  g0[2] = (unsigned)(ga & 0xFFFFFFFFull);       // global_addr[31:0]
  g0[3] = (unsigned)((ga >> 32) & 0x1FFFFFFull) // global_addr[56:32]
          | (2u << 30);                         // type=2 (image)
  v8i g1;
  // data_size=1 (2B) | pad_enable | pad_interval=3 (16 DWORDs) | pad_amount=3 (4 DWORDs)
  g1[0] = (int)((1u << 16) | (1u << 20) | (3u << 22) | (3u << 25));
  g1[1] = (int)(((unsigned)K & 0xFFFFu) << 16);               // tensor_dim0 lo16
  g1[2] = (int)((((unsigned)K >> 16) & 0xFFFFu) |             // tensor_dim0 hi16
                (((unsigned)tensor_rows & 0xFFFFu) << 16));   // tensor_dim1 lo16
  g1[3] = (int)((((unsigned)tensor_rows >> 16) & 0xFFFFu) |   // tensor_dim1 hi16
                (32u << 16));                                 // tile_dim0 = 32
  g1[4] = 128;                                                // tile_dim1 = 128
  g1[5] = K;                                                  // dim0_stride lo32
  g1[6] = 0;
  g1[7] = 0;
  const v4i z4 = {0, 0, 0, 0};
  const v8i z8 = {0, 0, 0, 0, 0, 0, 0, 0};
  __builtin_amdgcn_tensor_load_to_lds(g0, g1, z4, z4, z8, 0);
}

// ---------------------------------------------------------------------------
// Tiled bf16 WMMA GEMM:  out = epilogue(A[M,K] @ Bt[N,K]^T + bias [, res])
//   EPI 0: out bf16 = acc + bias
//   EPI 1: out bf16 = gelu_exact(acc + bias)
//   EPI 2: out f32  = res + acc + bias
// 256 thr (8 wave32). Block tile 128x128, K-step 32, TDM double-buffered LDS.
// Wave grid 2(M) x 4(N); each wave: 4x2 tiles of 16x16x32 wmma -> 64x32.
// ---------------------------------------------------------------------------
template <int EPI>
__global__ __launch_bounds__(256) void gemm_bf16_kernel(
    const __bf16* __restrict__ A, const __bf16* __restrict__ Bt,
    const float* __restrict__ bias, const float* __restrict__ res,
    void* __restrict__ outv, int M, int N, int K) {
  __shared__ __bf16 As[2][BM * LDT];
  __shared__ __bf16 Bs[2][BN * LDT];

  const int tid = threadIdx.x;
  const int lane = tid & 31;
  const int wave = tid >> 5;
  const int waveM = wave & 1;   // 0..1
  const int waveN = wave >> 1;  // 0..3
  const int bm = blockIdx.y * BM;
  const int bn = blockIdx.x * BN;
  const bool issuer = (wave == 0);

  const int halfK = (lane >> 4) * 8;  // 0 or 8
  const int r16 = lane & 15;

  v8f acc[4][2];
  const v8f vzero = {0.f, 0.f, 0.f, 0.f, 0.f, 0.f, 0.f, 0.f};
#pragma unroll
  for (int mt = 0; mt < 4; ++mt)
#pragma unroll
    for (int nt = 0; nt < 2; ++nt) acc[mt][nt] = vzero;

  // prologue: DMA first K-slab into buffer 0
  if (issuer) {
    tdm_load_tile((unsigned)(uintptr_t)&As[0][0], A + (size_t)bm * K, M, K);
    tdm_load_tile((unsigned)(uintptr_t)&Bs[0][0], Bt + (size_t)bn * K, N, K);
    __builtin_amdgcn_s_wait_tensorcnt(0);
  }
  __syncthreads();

  const int nsteps = K / BK;
  for (int ks = 0; ks < nsteps; ++ks) {
    const int cur = ks & 1;
    const int nxt = cur ^ 1;

    // kick off next slab's DMA before computing (overlap DMA with WMMA)
    if (issuer && (ks + 1 < nsteps)) {
      const int k1 = (ks + 1) * BK;
      tdm_load_tile((unsigned)(uintptr_t)&As[nxt][0],
                    A + (size_t)bm * K + k1, M, K);
      tdm_load_tile((unsigned)(uintptr_t)&Bs[nxt][0],
                    Bt + (size_t)bn * K + k1, N, K);
    }

    // fragments (A and B^T are both row-major, K-contiguous, LDT stride)
    v16bf af[4], bfr[2];
#pragma unroll
    for (int mt = 0; mt < 4; ++mt) {
      const __bf16* p = &As[cur][(waveM * 64 + mt * 16 + r16) * LDT];
      v16bf f;
#pragma unroll
      for (int i = 0; i < 8; ++i) {
        f[i] = p[halfK + i];
        f[8 + i] = p[16 + halfK + i];
      }
      af[mt] = f;
    }
#pragma unroll
    for (int nt = 0; nt < 2; ++nt) {
      const __bf16* p = &Bs[cur][(waveN * 32 + nt * 16 + r16) * LDT];
      v16bf f;
#pragma unroll
      for (int i = 0; i < 8; ++i) {
        f[i] = p[halfK + i];
        f[8 + i] = p[16 + halfK + i];
      }
      bfr[nt] = f;
    }

#pragma unroll
    for (int mt = 0; mt < 4; ++mt)
#pragma unroll
      for (int nt = 0; nt < 2; ++nt)
        acc[mt][nt] = __builtin_amdgcn_wmma_f32_16x16x32_bf16(
            false, af[mt], false, bfr[nt], (short)0, acc[mt][nt], false, false);

    if (issuer) __builtin_amdgcn_s_wait_tensorcnt(0);
    __syncthreads();  // publish nxt buffer / retire cur buffer
  }

  // --- epilogue (C/D layout: VGPR j -> M = j + 8*(lane>=16), N = lane&15) ---
#pragma unroll
  for (int mt = 0; mt < 4; ++mt) {
#pragma unroll
    for (int nt = 0; nt < 2; ++nt) {
      const int row0 = bm + waveM * 64 + mt * 16 + (lane >> 4) * 8;
      const int col = bn + waveN * 32 + nt * 16 + (lane & 15);
      const float bia = bias[col];
#pragma unroll
      for (int j = 0; j < 8; ++j) {
        const size_t idx = (size_t)(row0 + j) * N + col;
        const float a = acc[mt][nt][j] + bia;
        if constexpr (EPI == 0) {
          ((__bf16*)outv)[idx] = (__bf16)a;
        } else if constexpr (EPI == 1) {
          const float ge = 0.5f * a * (1.0f + erff(a * 0.70710678118654752f));
          ((__bf16*)outv)[idx] = (__bf16)ge;
        } else {
          ((float*)outv)[idx] = res[idx] + a;
        }
      }
    }
  }
}

// ---------------------------------------------------------------------------
// Cross-batch attention (faithful to reference einsums), B=8, HD=64.
// One 64-thread block per (s, h). Writes output pre-permuted so the Wo GEMM
// reads it as the reference's (B,H,S,HD)->view(S,B,E) flat buffer.
// ---------------------------------------------------------------------------
__global__ __launch_bounds__(64) void attn_kernel(
    const __bf16* __restrict__ q, const __bf16* __restrict__ k,
    const __bf16* __restrict__ v, __bf16* __restrict__ out) {
  const int s = blockIdx.x;
  const int h = blockIdx.y;
  __shared__ float qs[B_DIM][HD_DIM];
  __shared__ float ks[B_DIM][HD_DIM];
  __shared__ float vs[B_DIM][HD_DIM];
  __shared__ float sc[B_DIM][B_DIM];

  const int d = threadIdx.x;  // 0..63
#pragma unroll
  for (int b = 0; b < B_DIM; ++b) {
    const size_t base = ((size_t)(s * B_DIM + b)) * E_DIM + h * HD_DIM + d;
    qs[b][d] = (float)q[base];
    ks[b][d] = (float)k[base];
    vs[b][d] = (float)v[base];
  }
  __syncthreads();

  {  // scores: thread t -> (b,c) pair
    const int b = d >> 3, c = d & 7;
    float a = 0.f;
#pragma unroll
    for (int i = 0; i < HD_DIM; ++i) a += qs[b][i] * ks[c][i];
    sc[b][c] = a * 0.125f;  // HD^-0.5
  }
  __syncthreads();

  if (d < B_DIM) {  // softmax over c for row b=d
    float m = sc[d][0];
#pragma unroll
    for (int c = 1; c < B_DIM; ++c) m = fmaxf(m, sc[d][c]);
    float e[B_DIM], sum = 0.f;
#pragma unroll
    for (int c = 0; c < B_DIM; ++c) { e[c] = __expf(sc[d][c] - m); sum += e[c]; }
    const float inv = 1.0f / sum;
#pragma unroll
    for (int c = 0; c < B_DIM; ++c) sc[d][c] = e[c] * inv;
  }
  __syncthreads();

#pragma unroll
  for (int b = 0; b < B_DIM; ++b) {
    float o = 0.f;
#pragma unroll
    for (int c = 0; c < B_DIM; ++c) o += sc[b][c] * vs[c][d];
    // (b,h,s,d) flat in (B,H,S,HD) contiguous order
    out[((size_t)(b * H_DIM + h) * S_DIM + s) * HD_DIM + d] = (__bf16)o;
  }
}

// ---------------------------------------------------------------------------
// Launch
// ---------------------------------------------------------------------------
extern "C" void kernel_launch(void* const* d_in, const int* in_sizes, int n_in,
                              void* d_out, int out_size, void* d_ws, size_t ws_size,
                              hipStream_t stream) {
  const float* src = (const float*)d_in[0];
  const float* Wq = (const float*)d_in[1];
  const float* bq = (const float*)d_in[2];
  const float* Wk = (const float*)d_in[3];
  const float* bk = (const float*)d_in[4];
  const float* Wv = (const float*)d_in[5];
  const float* bv = (const float*)d_in[6];
  const float* Wo = (const float*)d_in[7];
  const float* bo = (const float*)d_in[8];
  const float* W1 = (const float*)d_in[9];
  const float* b1 = (const float*)d_in[10];
  const float* W2 = (const float*)d_in[11];
  const float* b2 = (const float*)d_in[12];
  const float* g1 = (const float*)d_in[13];
  const float* be1 = (const float*)d_in[14];
  const float* g2 = (const float*)d_in[15];
  const float* be2 = (const float*)d_in[16];

  const size_t MB = 1024ull * 1024ull;
  char* ws = (char*)d_ws;
  __bf16* wqb = (__bf16*)(ws + 0 * MB);    // 2MB   Wq^T [E,E] bf16
  __bf16* wkb = (__bf16*)(ws + 2 * MB);    // 2MB   Wk^T
  __bf16* wvb = (__bf16*)(ws + 4 * MB);    // 2MB   Wv^T
  __bf16* wob = (__bf16*)(ws + 6 * MB);    // 2MB   Wo^T
  __bf16* w1b = (__bf16*)(ws + 8 * MB);    // 8MB   W1^T [FF,E]
  __bf16* w2b = (__bf16*)(ws + 16 * MB);   // 8MB   W2^T [E,FF]
  __bf16* hb  = (__bf16*)(ws + 24 * MB);   // 64MB  LN1 out / LN2 out
  __bf16* qb  = (__bf16*)(ws + 88 * MB);   // 64MB
  __bf16* kb  = (__bf16*)(ws + 152 * MB);  // 64MB
  __bf16* vb  = (__bf16*)(ws + 216 * MB);  // 64MB
  __bf16* ab  = (__bf16*)(ws + 280 * MB);  // 64MB attn out (permuted flat)
  float*  x1  = (float*)(ws + 344 * MB);   // 128MB residual-1 result
  __bf16* fb  = (__bf16*)(ws + 88 * MB);   // 256MB ffn mid (reuses q/k/v/attn)

  // 1) weights -> transposed bf16 (done once; B-tiles become K-contiguous)
  transpose_bf16_kernel<<<dim3(E_DIM / 32, E_DIM / 32), 256, 0, stream>>>(
      Wq, wqb, E_DIM, E_DIM);
  transpose_bf16_kernel<<<dim3(E_DIM / 32, E_DIM / 32), 256, 0, stream>>>(
      Wk, wkb, E_DIM, E_DIM);
  transpose_bf16_kernel<<<dim3(E_DIM / 32, E_DIM / 32), 256, 0, stream>>>(
      Wv, wvb, E_DIM, E_DIM);
  transpose_bf16_kernel<<<dim3(E_DIM / 32, E_DIM / 32), 256, 0, stream>>>(
      Wo, wob, E_DIM, E_DIM);
  transpose_bf16_kernel<<<dim3(FF_DIM / 32, E_DIM / 32), 256, 0, stream>>>(
      W1, w1b, E_DIM, FF_DIM);
  transpose_bf16_kernel<<<dim3(E_DIM / 32, FF_DIM / 32), 256, 0, stream>>>(
      W2, w2b, FF_DIM, E_DIM);

  // 2) LN1
  ln_bf16_kernel<<<MROWS, 256, 0, stream>>>(src, g1, be1, hb);

  // 3) Q/K/V projections
  dim3 gEE(E_DIM / BN, MROWS / BM);  // (8, 256)
  gemm_bf16_kernel<0><<<gEE, 256, 0, stream>>>(hb, wqb, bq, nullptr, qb,
                                               MROWS, E_DIM, E_DIM);
  gemm_bf16_kernel<0><<<gEE, 256, 0, stream>>>(hb, wkb, bk, nullptr, kb,
                                               MROWS, E_DIM, E_DIM);
  gemm_bf16_kernel<0><<<gEE, 256, 0, stream>>>(hb, wvb, bv, nullptr, vb,
                                               MROWS, E_DIM, E_DIM);

  // 4) cross-batch attention
  dim3 gAttn(S_DIM, H_DIM);
  attn_kernel<<<gAttn, 64, 0, stream>>>(qb, kb, vb, ab);

  // 5) output projection + residual -> x1 (f32)
  gemm_bf16_kernel<2><<<gEE, 256, 0, stream>>>(ab, wob, bo, src, x1,
                                               MROWS, E_DIM, E_DIM);

  // 6) LN2
  ln_bf16_kernel<<<MROWS, 256, 0, stream>>>(x1, g2, be2, hb);

  // 7) FFN up + exact GELU -> fb (bf16)
  dim3 gFF1(FF_DIM / BN, MROWS / BM);  // (32, 256)
  gemm_bf16_kernel<1><<<gFF1, 256, 0, stream>>>(hb, w1b, b1, nullptr, fb,
                                                MROWS, FF_DIM, E_DIM);

  // 8) FFN down + residual -> d_out (f32)
  gemm_bf16_kernel<2><<<gEE, 256, 0, stream>>>(fb, w2b, b2, x1, (float*)d_out,
                                               MROWS, E_DIM, FF_DIM);
}
